// PCVRDeepContextNet_42786464203268
// MI455X (gfx1250) — compile-verified, matrix operand order checked by reading, not run on hardware
//
#include <hip/hip_runtime.h>
#include <math.h>

// ---------------- model dims ----------------
constexpr int kD   = 768;            // d_model
constexpr int kH   = 12;             // heads
constexpr int kDh  = kD / kH;        // 64
constexpr int kL   = 2;              // blocks
constexpr int kN   = 3;              // domains
constexpr int kB   = 64;             // batch
constexpr int kS   = 512;            // seq len
constexpr int kHid = 4 * kD;         // 3072
constexpr int kA   = 1;              // actions
constexpr int kHP  = 16;             // heads padded to WMMA N=16

// ---------------- WMMA types (CDNA5 wave32) ----------------
typedef __attribute__((ext_vector_type(16))) __bf16 v16bf;
typedef __attribute__((ext_vector_type(8)))  float  v8f;

union FragBF {
  v16bf v;
  unsigned short h[16];
  uint4 q[2];
};

static __device__ __forceinline__ unsigned short f2bf(float f) {
  unsigned int u = __float_as_uint(f);
  unsigned int r = u + 0x7fffu + ((u >> 16) & 1u);   // round-to-nearest-even
  return (unsigned short)(r >> 16);
}

// A / B fragment, 16x32 bf16, source row-major with leading dim `ld` (elements).
// Lane L: m = L%16 (row for A / N-col for B), g = L/16.
// elements 0..7  -> K = 8g .. 8g+7        (one b128)
// elements 8..15 -> K = 16+8g .. 16+8g+7  (one b128)
static __device__ __forceinline__ v16bf frag_ld_rm(const unsigned short* p, int ld, int lane) {
  int m = lane & 15, g = lane >> 4;
  const unsigned short* r = p + (size_t)m * ld + 8 * g;
  FragBF f;
  f.q[0] = *(const uint4*)(r);
  f.q[1] = *(const uint4*)(r + 16);
  return f.v;
}

static __device__ __forceinline__ v8f wmma_bf16(v16bf a, v16bf b, v8f c) {
  // (neg_a, A, neg_b, B, c_mod, C, reuse_a, reuse_b)
  return __builtin_amdgcn_wmma_f32_16x16x32_bf16(false, a, false, b, (short)0, c, false, false);
}

// ---------------- block-wide reduction (wave32) ----------------
static __device__ __forceinline__ float block_sum(float v, float* buf) {
#pragma unroll
  for (int off = 16; off; off >>= 1) v += __shfl_xor(v, off, 32);
  int tid = threadIdx.x;
  int w = tid >> 5;
  int nw = blockDim.x >> 5;
  if ((tid & 31) == 0) buf[w] = v;
  __syncthreads();
  float r = 0.f;
  if (w == 0) {
    r = (tid < nw) ? buf[tid] : 0.f;
#pragma unroll
    for (int off = 8; off; off >>= 1) r += __shfl_xor(r, off, 32);
    if (tid == 0) buf[0] = r;
  }
  __syncthreads();
  r = buf[0];
  __syncthreads();
  return r;
}

// ---------------- kernels ----------------

__global__ void k_copy(const float* __restrict__ src, float* __restrict__ dst, int n) {
  int i = blockIdx.x * blockDim.x + threadIdx.x;
  if (i < n) dst[i] = src[i];
}

// Per-token LN statistics of sequences -> x-hat in bf16 (token-major).  grid = N*B*S, block = 256
__global__ void k_xhat(const float* __restrict__ seq, unsigned short* __restrict__ xhat) {
  __shared__ float red[16];
  size_t t = blockIdx.x;
  const float* x = seq + t * kD;
  int tid = threadIdx.x;
  float v0 = x[tid], v1 = x[tid + 256], v2 = x[tid + 512];
  float s  = block_sum(v0 + v1 + v2, red);
  float ss = block_sum(v0 * v0 + v1 * v1 + v2 * v2, red);
  float m = s * (1.0f / kD);
  float var = ss * (1.0f / kD) - m * m;
  float rstd = rsqrtf(var + 1e-5f);
  unsigned short* o = xhat + t * kD;
  o[tid]       = f2bf((v0 - m) * rstd);
  o[tid + 256] = f2bf((v1 - m) * rstd);
  o[tid + 512] = f2bf((v2 - m) * rstd);
}

// Transpose x-hat within each (n,b): [S][D] -> [D][S], 64x64 tiles via padded LDS.
// grid = (N*B, S/64 * D/64), block = 256.  Coalesced on both sides.
__global__ void k_transpose(const unsigned short* __restrict__ xhat,
                            unsigned short* __restrict__ xhatT) {
  __shared__ unsigned short tile[64][66];              // +2 pad: no bank conflicts
  int nb = blockIdx.x;
  int t = blockIdx.y;
  int s0 = (t % (kS / 64)) * 64;
  int d0 = (t / (kS / 64)) * 64;
  int tid = threadIdx.x;
  const unsigned short* src = xhat + ((size_t)nb * kS + s0) * kD + d0;
#pragma unroll
  for (int i = 0; i < 16; ++i) {
    int idx = tid + 256 * i;                           // idx = r*64 + c  (r = s-local, c = d-local)
    int r = idx >> 6, c = idx & 63;
    tile[r][c] = src[(size_t)r * kD + c];
  }
  __syncthreads();
  unsigned short* dst = xhatT + ((size_t)nb * kD + d0) * kS + s0;
#pragma unroll
  for (int i = 0; i < 16; ++i) {
    int idx = tid + 256 * i;                           // idx = r2*64 + c2 (r2 = d-local, c2 = s-local)
    int r2 = idx >> 6, c2 = idx & 63;
    dst[(size_t)r2 * kS + c2] = tile[c2][r2];
  }
}

// q = LN(ctx); qh = q@Wq^T + bq; u[b,h,d] = scale * sn_w[d] * sum_e qh[h,e]*Wk[h*64+e,d]
// grid = B, block = 256.  u stored [B,16,768] bf16 (heads >= 12 zeroed).
__global__ void k_qu(const float* __restrict__ ctx,
                     const float* __restrict__ qn_w, const float* __restrict__ qn_b,
                     const float* __restrict__ sn_w,
                     const float* __restrict__ in_w, const float* __restrict__ in_b,
                     unsigned short* __restrict__ u) {
  __shared__ float qln[kD];
  __shared__ float qh[kD];
  __shared__ float red[16];
  int b = blockIdx.x, tid = threadIdx.x;
  const float* x = ctx + (size_t)b * kD;
  float v0 = x[tid], v1 = x[tid + 256], v2 = x[tid + 512];
  float s  = block_sum(v0 + v1 + v2, red);
  float ss = block_sum(v0 * v0 + v1 * v1 + v2 * v2, red);
  float m = s * (1.0f / kD);
  float rstd = rsqrtf(ss * (1.0f / kD) - m * m + 1e-5f);
  qln[tid]       = (v0 - m) * rstd * qn_w[tid]       + qn_b[tid];
  qln[tid + 256] = (v1 - m) * rstd * qn_w[tid + 256] + qn_b[tid + 256];
  qln[tid + 512] = (v2 - m) * rstd * qn_w[tid + 512] + qn_b[tid + 512];
  __syncthreads();
  for (int j = tid; j < kD; j += 256) {
    const float* w = in_w + (size_t)j * kD;            // Wq row j
    float acc = in_b[j];
    for (int d = 0; d < kD; ++d) acc += qln[d] * w[d];
    qh[j] = acc;
  }
  __syncthreads();
  const float scale = 0.125f;                          // 1/sqrt(64)
  for (int idx = tid; idx < kHP * kD; idx += 256) {
    int h = idx / kD, d = idx - h * kD;
    float acc = 0.f;
    if (h < kH) {
      const float* wk = in_w + ((size_t)(kD + h * kDh)) * kD + d;  // Wk column d, rows of head h
      const float* qv = qh + h * kDh;
      for (int e = 0; e < kDh; ++e) acc += qv[e] * wk[(size_t)e * kD];
      acc *= scale * sn_w[d];
    }
    u[(size_t)b * kHP * kD + idx] = f2bf(acc);
  }
}

// scores[nb][s][hp] = xhat[nb] (SxD) @ u[b]^T (Dx16).  WMMA bf16.
// grid = (N*B, 4), block = 256 (8 waves, 1 row-tile each).
__global__ void __launch_bounds__(256) k_scores(const unsigned short* __restrict__ xhat,
                                                const unsigned short* __restrict__ u,
                                                float* __restrict__ scores) {
  int nb = blockIdx.x;
  int b = nb % kB;
  int wave = threadIdx.x >> 5, lane = threadIdx.x & 31;
  int tile = blockIdx.y * 8 + wave;                    // 0..31 tiles of 16 rows over S
  const unsigned short* xb = xhat + (size_t)nb * kS * kD + (size_t)(tile * 16) * kD;
  const unsigned short* ub = u + (size_t)b * kHP * kD; // [16][768], N-major => fast B path
  v8f acc = {};
  for (int k0 = 0; k0 < kD; k0 += 32) {
    v16bf a  = frag_ld_rm(xb + k0, kD, lane);
    v16bf bm = frag_ld_rm(ub + k0, kD, lane);
    acc = wmma_bf16(a, bm, acc);
  }
  int n = lane & 15, g = lane >> 4;
  float* out = scores + ((size_t)nb * kS + tile * 16) * kHP;
#pragma unroll
  for (int r = 0; r < 8; ++r) out[(r + 8 * g) * kHP + n] = acc[r];
}

// masked softmax per (n,b,h); attn stored bf16 [nb][16][512].  grid = N*B, block = 512
__global__ void k_softmax(const float* __restrict__ scores, const int* __restrict__ seq_lens,
                          unsigned short* __restrict__ attn) {
  int nb = blockIdx.x;
  int dn = nb / kB, b = nb % kB;
  int h = threadIdx.x >> 5, lane = threadIdx.x & 31;
  int len = seq_lens[dn * kB + b];
  const float* sc = scores + (size_t)nb * kS * kHP;
  float v[16];
  float mx = -3.0e38f;
#pragma unroll
  for (int i = 0; i < 16; ++i) {
    int s = lane + 32 * i;
    float x = (s < len) ? sc[(size_t)s * kHP + h] : -3.0e38f;
    v[i] = x;
    mx = fmaxf(mx, x);
  }
#pragma unroll
  for (int off = 16; off; off >>= 1) mx = fmaxf(mx, __shfl_xor(mx, off, 32));
  float sum = 0.f;
#pragma unroll
  for (int i = 0; i < 16; ++i) {
    int s = lane + 32 * i;
    float e = (s < len) ? __expf(v[i] - mx) : 0.f;
    v[i] = e;
    sum += e;
  }
#pragma unroll
  for (int off = 16; off; off >>= 1) sum += __shfl_xor(sum, off, 32);
  float inv = 1.0f / sum;
  unsigned short* ao = attn + ((size_t)nb * kHP + h) * kS;
#pragma unroll
  for (int i = 0; i < 16; ++i) ao[lane + 32 * i] = f2bf(v[i] * inv);
}

// P[nb][16][768] = attn (16xS) @ xhat (SxD).  WMMA bf16; B now from d-major xhatT => b128 path.
// grid = (N*B, 2), block = 256 (8 waves x 3 N-tiles each).
__global__ void __launch_bounds__(256) k_pool(const unsigned short* __restrict__ xhatT,
                                              const unsigned short* __restrict__ attn,
                                              float* __restrict__ P) {
  int nb = blockIdx.x;
  int wave = threadIdx.x >> 5, lane = threadIdx.x & 31;
  const unsigned short* xbT = xhatT + (size_t)nb * kD * kS;  // [768][512], N-major over d
  const unsigned short* ab  = attn + (size_t)nb * kHP * kS;
  v8f z = {};
  v8f acc[3];
  acc[0] = z; acc[1] = z; acc[2] = z;
  int tbase = blockIdx.y * 24 + wave * 3;              // 48 tiles of 16 over D
  for (int k0 = 0; k0 < kS; k0 += 32) {
    v16bf a = frag_ld_rm(ab + k0, kS, lane);           // A reused for 3 WMMAs
#pragma unroll
    for (int j = 0; j < 3; ++j) {
      int n0 = (tbase + j) * 16;
      v16bf bm = frag_ld_rm(xbT + (size_t)n0 * kS + k0, kS, lane);
      acc[j] = wmma_bf16(a, bm, acc[j]);
    }
  }
  int n = lane & 15, g = lane >> 4;
  float* po = P + (size_t)nb * kHP * kD;
#pragma unroll
  for (int j = 0; j < 3; ++j) {
    int n0 = (tbase + j) * 16;
#pragma unroll
    for (int r = 0; r < 8; ++r) po[(size_t)(r + 8 * g) * kD + n0 + n] = acc[j][r];
  }
}

// kvbar = sn_w*P + sn_b; ctx[h,e] = kvbar[h]·Wv_row + bv; attended = ctx@out_w^T + out_b.
// grid = N*B, block = 256.
__global__ void k_attended(const float* __restrict__ P,
                           const float* __restrict__ sn_w, const float* __restrict__ sn_b,
                           const float* __restrict__ in_w, const float* __restrict__ in_b,
                           const float* __restrict__ out_w, const float* __restrict__ out_b,
                           float* __restrict__ att) {
  __shared__ float kvbar[kH * kD];                     // 36 KB
  __shared__ float cvec[kD];
  int nb = blockIdx.x, tid = threadIdx.x;
  const float* Pb = P + (size_t)nb * kHP * kD;
  for (int idx = tid; idx < kH * kD; idx += 256) {
    int d = idx % kD;
    kvbar[idx] = sn_w[d] * Pb[idx] + sn_b[d];
  }
  __syncthreads();
  for (int idx = tid; idx < kD; idx += 256) {          // idx = h*64 + e
    int h = idx >> 6;
    int row = 2 * kD + idx;                            // Wv row
    const float* w = in_w + (size_t)row * kD;
    const float* kv = kvbar + h * kD;
    float acc = in_b[row];
    for (int d = 0; d < kD; ++d) acc += kv[d] * w[d];
    cvec[idx] = acc;
  }
  __syncthreads();
  float* ao = att + (size_t)nb * kD;
  for (int j = tid; j < kD; j += 256) {
    const float* w = out_w + (size_t)j * kD;
    float acc = out_b[j];
    for (int i = 0; i < kD; ++i) acc += cvec[i] * w[i];
    ao[j] = acc;
  }
}

// seq_ctx = mean_n attended; gate; residual; LN; FFN (SiLU); residual.  In-place ctx row.
// grid = B, block = 256.
__global__ void k_gate_ffn(const float* __restrict__ att,
                           const float* __restrict__ gate_w, const float* __restrict__ gate_b,
                           const float* __restrict__ ln_w, const float* __restrict__ ln_b,
                           const float* __restrict__ w1, const float* __restrict__ b1,
                           const float* __restrict__ w2, const float* __restrict__ b2,
                           float* __restrict__ ctx) {
  __shared__ float sctx[kD];
  __shared__ float cold[kD];
  __shared__ float cnew[kD];
  __shared__ float hln[kD];
  __shared__ float t1[kHid];                           // 12 KB
  __shared__ float red[16];
  int b = blockIdx.x, tid = threadIdx.x;
  float* cb = ctx + (size_t)b * kD;
  for (int d = tid; d < kD; d += 256) {
    float v = 0.f;
    for (int n = 0; n < kN; ++n) v += att[((size_t)n * kB + b) * kD + d];
    sctx[d] = v * (1.0f / kN);
    cold[d] = cb[d];
  }
  __syncthreads();
  for (int j = tid; j < kD; j += 256) {
    const float* w = gate_w + (size_t)j * (2 * kD);
    float acc = gate_b[j];
    for (int d = 0; d < kD; ++d) acc += w[d] * cold[d];
    for (int d = 0; d < kD; ++d) acc += w[kD + d] * sctx[d];
    float g = 1.0f / (1.0f + __expf(-acc));
    cnew[j] = cold[j] + g * sctx[j];
  }
  // block_sum's first barrier orders all cnew writes before hln writes below
  float v0 = cnew[tid], v1 = cnew[tid + 256], v2 = cnew[tid + 512];
  float s  = block_sum(v0 + v1 + v2, red);
  float ss = block_sum(v0 * v0 + v1 * v1 + v2 * v2, red);
  float m = s * (1.0f / kD);
  float rstd = rsqrtf(ss * (1.0f / kD) - m * m + 1e-5f);
  hln[tid]       = (v0 - m) * rstd * ln_w[tid]       + ln_b[tid];
  hln[tid + 256] = (v1 - m) * rstd * ln_w[tid + 256] + ln_b[tid + 256];
  hln[tid + 512] = (v2 - m) * rstd * ln_w[tid + 512] + ln_b[tid + 512];
  __syncthreads();
  for (int j = tid; j < kHid; j += 256) {
    const float* w = w1 + (size_t)j * kD;
    float acc = b1[j];
    for (int d = 0; d < kD; ++d) acc += hln[d] * w[d];
    t1[j] = acc / (1.0f + __expf(-acc));               // SiLU
  }
  __syncthreads();
  for (int d = tid; d < kD; d += 256) {
    const float* w = w2 + (size_t)d * kHid;
    float acc = b2[d];
    for (int i = 0; i < kHid; ++i) acc += t1[i] * w[i];
    cb[d] = cnew[d] + acc;
  }
}

// emb = LN(ctx, out_norm); h = LN(emb, cls_ln); out = silu(h@W1^T+b1)@W2^T + b2
// grid = B, block = 256.
__global__ void k_head(const float* __restrict__ ctx,
                       const float* __restrict__ on_w, const float* __restrict__ on_b,
                       const float* __restrict__ cl_w, const float* __restrict__ cl_b,
                       const float* __restrict__ w1, const float* __restrict__ b1,
                       const float* __restrict__ w2, const float* __restrict__ b2,
                       float* __restrict__ out) {
  __shared__ float h1[kD];
  __shared__ float t[kHid];
  __shared__ float red[16];
  int b = blockIdx.x, tid = threadIdx.x;
  const float* x = ctx + (size_t)b * kD;
  float v0 = x[tid], v1 = x[tid + 256], v2 = x[tid + 512];
  float s  = block_sum(v0 + v1 + v2, red);
  float ss = block_sum(v0 * v0 + v1 * v1 + v2 * v2, red);
  float m = s * (1.0f / kD);
  float rstd = rsqrtf(ss * (1.0f / kD) - m * m + 1e-5f);
  float u0 = (v0 - m) * rstd * on_w[tid]       + on_b[tid];
  float u1 = (v1 - m) * rstd * on_w[tid + 256] + on_b[tid + 256];
  float u2 = (v2 - m) * rstd * on_w[tid + 512] + on_b[tid + 512];
  s  = block_sum(u0 + u1 + u2, red);
  ss = block_sum(u0 * u0 + u1 * u1 + u2 * u2, red);
  m = s * (1.0f / kD);
  rstd = rsqrtf(ss * (1.0f / kD) - m * m + 1e-5f);
  h1[tid]       = (u0 - m) * rstd * cl_w[tid]       + cl_b[tid];
  h1[tid + 256] = (u1 - m) * rstd * cl_w[tid + 256] + cl_b[tid + 256];
  h1[tid + 512] = (u2 - m) * rstd * cl_w[tid + 512] + cl_b[tid + 512];
  __syncthreads();
  for (int j = tid; j < kHid; j += 256) {
    const float* w = w1 + (size_t)j * kD;
    float acc = b1[j];
    for (int d = 0; d < kD; ++d) acc += h1[d] * w[d];
    t[j] = acc / (1.0f + __expf(-acc));
  }
  __syncthreads();
  for (int a = 0; a < kA; ++a) {
    float acc = 0.f;
    for (int j = tid; j < kHid; j += 256) acc += t[j] * w2[(size_t)a * kHid + j];
    acc = block_sum(acc, red);
    if (tid == 0) out[b * kA + a] = acc + b2[a];
  }
}

// ---------------- host launcher ----------------
extern "C" void kernel_launch(void* const* d_in, const int* in_sizes, int n_in,
                              void* d_out, int out_size, void* d_ws, size_t ws_size,
                              hipStream_t stream) {
  (void)in_sizes; (void)n_in; (void)out_size; (void)ws_size;
  const float* context   = (const float*)d_in[0];
  const float* sequences = (const float*)d_in[1];
  const int*   seq_lens  = (const int*)d_in[2];
  const float* qn_w = (const float*)d_in[3];
  const float* qn_b = (const float*)d_in[4];
  const float* sn_w = (const float*)d_in[5];
  const float* sn_b = (const float*)d_in[6];
  const float* in_w = (const float*)d_in[7];
  const float* in_b = (const float*)d_in[8];
  const float* out_w = (const float*)d_in[9];
  const float* out_b = (const float*)d_in[10];
  const float* gate_w = (const float*)d_in[11];
  const float* gate_b = (const float*)d_in[12];
  const float* ffn_ln_w = (const float*)d_in[13];
  const float* ffn_ln_b = (const float*)d_in[14];
  const float* ffn_w1 = (const float*)d_in[15];
  const float* ffn_b1 = (const float*)d_in[16];
  const float* ffn_w2 = (const float*)d_in[17];
  const float* ffn_b2 = (const float*)d_in[18];
  const float* on_w = (const float*)d_in[19];
  const float* on_b = (const float*)d_in[20];
  const float* cl_w = (const float*)d_in[21];
  const float* cl_b = (const float*)d_in[22];
  const float* cw1 = (const float*)d_in[23];
  const float* cb1 = (const float*)d_in[24];
  const float* cw2 = (const float*)d_in[25];
  const float* cb2 = (const float*)d_in[26];
  float* out = (float*)d_out;

  char* wsb = (char*)d_ws;
  size_t cur = 0;
  auto take = [&](size_t bytes) -> void* {
    void* p = wsb + cur;
    cur = (cur + bytes + 255) & ~(size_t)255;
    return p;
  };
  unsigned short* xhat   = (unsigned short*)take((size_t)kN * kB * kS * kD * 2);   // 151 MB
  unsigned short* xhatT  = (unsigned short*)take((size_t)kN * kB * kD * kS * 2);   // 151 MB (d-major)
  unsigned short* ubuf   = (unsigned short*)take((size_t)kB * kHP * kD * 2);
  float*          scores = (float*)take((size_t)kN * kB * kS * kHP * 4);
  unsigned short* attn   = (unsigned short*)take((size_t)kN * kB * kHP * kS * 2);
  float*          P      = (float*)take((size_t)kN * kB * kHP * kD * 4);
  float*          att    = (float*)take((size_t)kN * kB * kD * 4);
  float*          ctx    = (float*)take((size_t)kB * kD * 4);

  k_copy<<<(kB * kD + 255) / 256, 256, 0, stream>>>(context, ctx, kB * kD);
  k_xhat<<<kN * kB * kS, 256, 0, stream>>>(sequences, xhat);
  k_transpose<<<dim3(kN * kB, (kS / 64) * (kD / 64)), 256, 0, stream>>>(xhat, xhatT);

  for (int l = 0; l < kL; ++l) {
    const float* inw_l = in_w + (size_t)l * 3 * kD * kD;
    const float* inb_l = in_b + (size_t)l * 3 * kD;
    k_qu<<<kB, 256, 0, stream>>>(ctx, qn_w + l * kD, qn_b + l * kD, sn_w + l * kD,
                                 inw_l, inb_l, ubuf);
    k_scores<<<dim3(kN * kB, 4), 256, 0, stream>>>(xhat, ubuf, scores);
    k_softmax<<<kN * kB, 512, 0, stream>>>(scores, seq_lens, attn);
    k_pool<<<dim3(kN * kB, 2), 256, 0, stream>>>(xhatT, attn, P);
    k_attended<<<kN * kB, 256, 0, stream>>>(P, sn_w + l * kD, sn_b + l * kD,
                                            inw_l, inb_l,
                                            out_w + (size_t)l * kD * kD, out_b + l * kD, att);
    k_gate_ffn<<<kB, 256, 0, stream>>>(att,
                                       gate_w + (size_t)l * kD * 2 * kD, gate_b + l * kD,
                                       ffn_ln_w + l * kD, ffn_ln_b + l * kD,
                                       ffn_w1 + (size_t)l * kHid * kD, ffn_b1 + (size_t)l * kHid,
                                       ffn_w2 + (size_t)l * kD * kHid, ffn_b2 + l * kD,
                                       ctx);
  }
  k_head<<<kB, 256, 0, stream>>>(ctx, on_w, on_b, cl_w, cl_b, cw1, cb1, cw2, cb2, out);
}
